// SeHG_bio_59571196396038
// MI455X (gfx1250) — matrix operand order repaired and checked by reading, not compiled, not run on hardware
//
#include <hip/hip_runtime.h>
#include <hip/hip_fp16.h>

typedef __attribute__((ext_vector_type(16))) _Float16 v16h;
typedef __attribute__((ext_vector_type(8)))  _Float16 v8h;
typedef __attribute__((ext_vector_type(2)))  _Float16 v2h;
typedef __attribute__((ext_vector_type(8)))  float    v8f;
using half_t = _Float16;

namespace cfg {
constexpr int ND = 8192, NZ = 4096, IN = 256, HID = 128, MID = 192;
constexpr int P = 4, M = 5, DEG = 32;
constexpr long ED = (long)ND * DEG;   // edges per metapath, drug
constexpr long EZ = (long)NZ * DEG;   // edges per metapath, disease
constexpr int KBIG = M * HID;         // 640

constexpr size_t ALN = 256;
constexpr size_t al(size_t x) { return (x + ALN - 1) / ALN * ALN; }

// ---- workspace layout (bytes) ----
constexpr size_t OFF_DEG_D = 0;
constexpr size_t OFF_DEG_Z = al(OFF_DEG_D + (size_t)P * ND * 4);
constexpr size_t OFF_UNION = al(OFF_DEG_Z + (size_t)P * NZ * 4);
constexpr size_t SZ_AGG32  = (size_t)P * (ND + NZ) * IN * 4;      // 50.3 MB
constexpr size_t OFF_AGG32_D = OFF_UNION;
constexpr size_t OFF_AGG32_Z = OFF_AGG32_D + (size_t)P * ND * IN * 4;
// --- union aliases: agg32 is dead once aggh is built; reuse for Q/K/attn-out ---
constexpr size_t OFF_Q_D   = OFF_UNION;
constexpr size_t OFF_K_D   = OFF_Q_D   + (size_t)M * ND * HID * 2;
constexpr size_t OFF_Q_Z   = OFF_K_D   + (size_t)M * ND * HID * 2;
constexpr size_t OFF_K_Z   = OFF_Q_Z   + (size_t)M * NZ * HID * 2;
constexpr size_t OFF_OUT_D = OFF_K_Z   + (size_t)M * NZ * HID * 2;
constexpr size_t OFF_OUT_Z = OFF_OUT_D + (size_t)M * ND * HID * 2;
static_assert(OFF_OUT_Z + (size_t)M * NZ * HID * 2 <= OFF_UNION + SZ_AGG32, "union overflow");

constexpr size_t OFF_AGGH_D = al(OFF_UNION + SZ_AGG32);
constexpr size_t OFF_AGGH_Z = al(OFF_AGGH_D + (size_t)M * ND * IN * 2);
constexpr size_t OFF_H1_D   = al(OFF_AGGH_Z + (size_t)M * NZ * IN * 2);
constexpr size_t OFF_H1_Z   = al(OFF_H1_D + (size_t)M * ND * MID * 2);
constexpr size_t OFF_H2_D   = al(OFF_H1_Z + (size_t)M * NZ * MID * 2);
constexpr size_t OFF_H2_Z   = al(OFF_H2_D + (size_t)M * ND * MID * 2);
constexpr size_t OFF_PROJ_D = al(OFF_H2_Z + (size_t)M * NZ * MID * 2);
constexpr size_t OFF_PROJ_Z = al(OFF_PROJ_D + (size_t)M * ND * HID * 2);
constexpr size_t OFF_ABIG   = al(OFF_PROJ_Z + (size_t)M * NZ * HID * 2);
constexpr size_t OFF_BBIG   = al(OFF_ABIG + (size_t)ND * KBIG * 2);
constexpr size_t OFF_W1T_D  = al(OFF_BBIG + (size_t)NZ * KBIG * 2);
constexpr size_t OFF_W2T_D  = OFF_W1T_D + (size_t)M * IN * MID * 2;
constexpr size_t OFF_W3T_D  = OFF_W2T_D + (size_t)M * MID * MID * 2;
constexpr size_t OFF_W1T_Z  = OFF_W3T_D + (size_t)M * MID * HID * 2;
constexpr size_t OFF_W2T_Z  = OFF_W1T_Z + (size_t)M * IN * MID * 2;
constexpr size_t OFF_W3T_Z  = OFF_W2T_Z + (size_t)M * MID * MID * 2;
constexpr size_t OFF_WQT_D  = OFF_W3T_Z + (size_t)M * MID * HID * 2;
constexpr size_t OFF_WKT_D  = OFF_WQT_D + (size_t)HID * HID * 2;
constexpr size_t OFF_WQT_Z  = OFF_WKT_D + (size_t)HID * HID * 2;
constexpr size_t OFF_WKT_Z  = OFF_WQT_Z + (size_t)HID * HID * 2;
constexpr size_t OFF_WDECT  = OFF_WKT_Z + (size_t)HID * HID * 2;
} // namespace cfg

using namespace cfg;

// ---------------------------------------------------------------------------
// SPMM: degree count (atomic), reciprocal, edge scatter (atomic f32 adds)
// ---------------------------------------------------------------------------
__global__ void k_deg(const int* __restrict__ src, float* __restrict__ deg,
                      long totalPE, int E, int n) {
  long t = (long)blockIdx.x * blockDim.x + threadIdx.x;
  if (t >= totalPE) return;
  int p = (int)(t / E);
  int s = src[t];
  atomicAdd(&deg[(long)p * n + s], 1.0f);
}

__global__ void k_inv(float* __restrict__ deg, long total) {
  long t = (long)blockIdx.x * blockDim.x + threadIdx.x;
  if (t < total) deg[t] = 1.0f / fmaxf(deg[t], 1.0f);
}

// agg[p][s][:] += inv[p][s] * att[p][d] * feat[d][:]   (thread = edge x 64-feat chunk)
__global__ void k_scatter(const int* __restrict__ src, const int* __restrict__ dst,
                          const float* __restrict__ att, const float* __restrict__ feat,
                          const float* __restrict__ inv, float* __restrict__ agg,
                          int E, int n) {
  long t = (long)blockIdx.x * blockDim.x + threadIdx.x;
  long total = (long)P * E * 4;
  if (t >= total) return;
  int chunk = (int)(t & 3);
  long eg = t >> 2;
  int p = (int)(eg / E);
  long e = eg - (long)p * E;
  int s = src[(long)p * E + e];
  int d = dst[(long)p * E + e];
  float coef = inv[(long)p * n + s] * att[(long)p * n + d];
  const float* f = feat + (long)d * IN + chunk * 64;
  float* a = agg + ((long)p * n + s) * IN + chunk * 64;
#pragma unroll 8
  for (int i = 0; i < 64; i++) atomicAdd(&a[i], coef * f[i]);
}

// aggh[m][i][k] = f16( m==0 ? feat : agg[m-1] )
__global__ void k_aggh(const float* __restrict__ feat, const float* __restrict__ agg,
                       half_t* __restrict__ out, int n) {
  long t = (long)blockIdx.x * blockDim.x + threadIdx.x;
  long per = (long)n * IN;
  long total = (long)M * per;
  if (t >= total) return;
  int m = (int)(t / per);
  long rem = t - (long)m * per;
  float v = (m == 0) ? feat[rem] : agg[(long)(m - 1) * per + rem];
  out[t] = (half_t)v;
}

// Wt[c][col][row] = f16(W[c][row][col])   (W: (C,Kd,Nd) f32 -> Wt: (C,Nd,Kd) f16)
__global__ void k_wT(const float* __restrict__ W, half_t* __restrict__ Wt,
                     int C, int Kd, int Nd) {
  long t = (long)blockIdx.x * blockDim.x + threadIdx.x;
  long per = (long)Kd * Nd;
  if (t >= (long)C * per) return;
  int c = (int)(t / per);
  long rem = t - (long)c * per;
  int r = (int)(rem / Nd);
  int col = (int)(rem - (long)r * Nd);
  Wt[(long)c * per + (long)col * Kd + r] = (half_t)W[t];
}

// ---------------------------------------------------------------------------
// LDS-staged, double-buffered, software-pipelined f16 WMMA GEMM:
//   out = act( A(rows,K) @ Bt(N,K)^T + bias )
// Block = 256 threads (8 waves), block tile = 128 x BN, K-step = 32.
// Per step i: issue global loads for step i+1, compute FM*FN (4 or 8) WMMAs per wave for step i from
// LDS buffer i&1, stage step i+1 into buffer (i+1)&1, one barrier. Global-load
// latency hides under the WMMA burst. All control flow is workgroup-uniform so
// EXEC stays all-ones (WMMA requirement). Grid = (N/BN, rows/128, channels);
// all dims divide exactly. out index = ch*Och + row*ldOut + col.
// ---------------------------------------------------------------------------
template <int BN, int WMW, int WNW, bool RELU, bool OUT16>
__global__ __launch_bounds__(256) void k_gemm(
    const half_t* __restrict__ A, const half_t* __restrict__ Bt,
    const float* __restrict__ bias, void* __restrict__ outv,
    int K, int ldOut, long Ach, long Bch, long biasCh, long Och) {
  constexpr int BM = 128, BK = 32, LD = BK + 8;      // pad 8 halves (16 B)
  constexpr int FM = BM / (16 * WMW);
  constexpr int FN = BN / (16 * WNW);
  constexpr int AV = BM * BK / 256 / 8;              // v8h loads/thread for A (=2)
  constexpr int BV = BN * BK / 256 / 8;              // v8h loads/thread for B (2 or 1)
  __shared__ __align__(16) half_t sA[2][BM * LD];
  __shared__ __align__(16) half_t sB[2][BN * LD];

  int ch = blockIdx.z;
  A += (long)ch * Ach;
  Bt += (long)ch * Bch;
  const float* bptr = bias ? bias + (long)ch * biasCh : nullptr;

  int row0 = blockIdx.y * BM;
  int col0 = blockIdx.x * BN;
  int t = threadIdx.x;
  int wave = t >> 5, lane = t & 31;
  int hi = lane >> 4, lp = lane & 15;
  int wm = wave / WNW;
  int wn = wave % WNW;

  v8f zf = {};
  v8f c[FM][FN];
#pragma unroll
  for (int f = 0; f < FM; f++)
#pragma unroll
    for (int g = 0; g < FN; g++) c[f][g] = zf;

  int steps = K / BK;
  v8h ar[AV], brr[BV];

  // --- prologue: load + stage K-step 0 into buffer 0 ---
#pragma unroll
  for (int j = 0; j < AV; j++) {
    int flat = t * (AV * 8) + j * 8;
    int r = flat >> 5, o = flat & 31;
    ar[j] = *(const v8h*)(A + (long)(row0 + r) * K + o);
  }
#pragma unroll
  for (int j = 0; j < BV; j++) {
    int flat = t * (BV * 8) + j * 8;
    int r = flat >> 5, o = flat & 31;
    brr[j] = *(const v8h*)(Bt + (long)(col0 + r) * K + o);
  }
#pragma unroll
  for (int j = 0; j < AV; j++) {
    int flat = t * (AV * 8) + j * 8;
    int r = flat >> 5, o = flat & 31;
    *(v8h*)(sA[0] + r * LD + o) = ar[j];
  }
#pragma unroll
  for (int j = 0; j < BV; j++) {
    int flat = t * (BV * 8) + j * 8;
    int r = flat >> 5, o = flat & 31;
    *(v8h*)(sB[0] + r * LD + o) = brr[j];
  }
  __syncthreads();

  for (int i = 0; i < steps; i++) {
    bool more = (i + 1) < steps;
    int k0n = (i + 1) * BK;
    if (more) {                        // issue next step's global loads early
#pragma unroll
      for (int j = 0; j < AV; j++) {
        int flat = t * (AV * 8) + j * 8;
        int r = flat >> 5, o = flat & 31;
        ar[j] = *(const v8h*)(A + (long)(row0 + r) * K + k0n + o);
      }
#pragma unroll
      for (int j = 0; j < BV; j++) {
        int flat = t * (BV * 8) + j * 8;
        int r = flat >> 5, o = flat & 31;
        brr[j] = *(const v8h*)(Bt + (long)(col0 + r) * K + k0n + o);
      }
    }

    // --- compute step i from buffer i&1 (loads above fly underneath) ---
    const half_t* cA = sA[i & 1];
    const half_t* cB = sB[i & 1];
    v16h af[FM], bf[FN];
#pragma unroll
    for (int f = 0; f < FM; f++) {
      // 16-bit A 16x32 layout: VGPR v holds K = 2v(+8 if v>=4)(+8 for hi lanes), pairs
      const half_t* ap = cA + (wm * FM * 16 + f * 16 + lp) * LD;
#pragma unroll
      for (int v = 0; v < 8; v++) {
        int kk = 2 * v + (v >= 4 ? 8 : 0) + hi * 8;
        v2h pr = *(const v2h*)(ap + kk);
        af[f][2 * v]     = pr[0];
        af[f][2 * v + 1] = pr[1];
      }
    }
#pragma unroll
    for (int g = 0; g < FN; g++) {
      // 16-bit B 32x16 layout: elems 0-7 hold K=8*hi+e, elems 8-15 hold K=16+8*hi+e (matches A)
      const half_t* bp = cB + (wn * FN * 16 + g * 16 + lp) * LD + hi * 8;
      v8h b0 = *(const v8h*)bp;
      v8h b1 = *(const v8h*)(bp + 16);
#pragma unroll
      for (int e = 0; e < 8; e++) { bf[g][e] = b0[e]; bf[g][8 + e] = b1[e]; }
    }
#pragma unroll
    for (int f = 0; f < FM; f++)
#pragma unroll
      for (int g = 0; g < FN; g++)
        c[f][g] = __builtin_amdgcn_wmma_f32_16x16x32_f16(false, af[f], false, bf[g],
                                                         (short)0, c[f][g], false, false);

    // --- stage step i+1 into the other buffer (its last readers finished
    //     before the previous barrier), then one barrier ---
    if (more) {
      half_t* nA = sA[(i + 1) & 1];
      half_t* nB = sB[(i + 1) & 1];
#pragma unroll
      for (int j = 0; j < AV; j++) {
        int flat = t * (AV * 8) + j * 8;
        int r = flat >> 5, o = flat & 31;
        *(v8h*)(nA + r * LD + o) = ar[j];
      }
#pragma unroll
      for (int j = 0; j < BV; j++) {
        int flat = t * (BV * 8) + j * 8;
        int r = flat >> 5, o = flat & 31;
        *(v8h*)(nB + r * LD + o) = brr[j];
      }
    }
    __syncthreads();
  }

#pragma unroll
  for (int f = 0; f < FM; f++) {
#pragma unroll
    for (int g = 0; g < FN; g++) {
      int cc = col0 + wn * FN * 16 + g * 16 + lp;
      float bv = bptr ? bptr[cc] : 0.0f;
#pragma unroll
      for (int r = 0; r < 8; r++) {
        // C/D layout: VGPR r -> M = r (+8 for hi lanes)
        int rr = row0 + wm * FM * 16 + f * 16 + r + 8 * hi;
        float v = c[f][g][r] + bv;
        if (RELU) v = fmaxf(v, 0.0f);
        long idx = (long)ch * Och + (long)rr * ldOut + cc;
        if constexpr (OUT16) ((half_t*)outv)[idx] = (half_t)v;
        else                 ((float*)outv)[idx]  = v;
      }
    }
  }
}

// ---------------------------------------------------------------------------
// Per-node metapath attention. Block = 128 threads (= HID), blockIdx.x = node.
// Writes out in (n, M, HID) flat order (faithful torch .view reinterpretation).
// ---------------------------------------------------------------------------
__global__ void k_attn(const half_t* __restrict__ proj, const half_t* __restrict__ Q,
                       const half_t* __restrict__ Kh, const float* __restrict__ beta,
                       half_t* __restrict__ outb, int n) {
  __shared__ float sp[M][HID], sq[M][HID], sk[M][HID];
  __shared__ float sc[M][M];
  int i = blockIdx.x;
  int t = threadIdx.x;
  for (int m = 0; m < M; m++) {
    long idx = ((long)m * n + i) * HID + t;
    sp[m][t] = (float)proj[idx];
    sq[m][t] = (float)Q[idx];
    sk[m][t] = (float)Kh[idx];
  }
  __syncthreads();
  if (t < M * M) {
    int m = t / M, j = t % M;
    float s = 0.0f;
    for (int h = 0; h < HID; h++) s += sq[m][h] * sk[j][h];
    sc[m][j] = s;
  }
  __syncthreads();
  if (t < M) {
    float mx = sc[t][0];
    for (int j = 1; j < M; j++) mx = fmaxf(mx, sc[t][j]);
    float sum = 0.0f;
    for (int j = 0; j < M; j++) { float e = __expf(sc[t][j] - mx); sc[t][j] = e; sum += e; }
    float r = 1.0f / sum;
    for (int j = 0; j < M; j++) sc[t][j] *= r;
  }
  __syncthreads();
  float b = beta[0];
  for (int m = 0; m < M; m++) {
    float acc = 0.0f;
    for (int j = 0; j < M; j++) acc += sc[m][j] * sp[j][t];
    outb[((long)i * M + m) * HID + t] = (half_t)(b * acc + sp[m][t]);
  }
}

// Abig[d, m*HID+h] = softmax(wattn)[m] * out_d_view[m,d,h]
__global__ void k_abig(const half_t* __restrict__ outd, const float* __restrict__ wattn,
                       half_t* __restrict__ Abig) {
  long t = (long)blockIdx.x * blockDim.x + threadIdx.x;
  long total = (long)ND * KBIG;
  if (t >= total) return;
  int k = (int)(t % KBIG);
  long d = t / KBIG;
  int m = k / HID;
  int h = k - m * HID;
  float mx = wattn[0];
  for (int j = 1; j < M; j++) mx = fmaxf(mx, wattn[j]);
  float sum = 0.0f;
  for (int j = 0; j < M; j++) sum += __expf(wattn[j] - mx);
  float w = __expf(wattn[m] - mx) / sum;
  Abig[t] = (half_t)(w * (float)outd[((long)m * ND + d) * HID + h]);
}

// ---------------------------------------------------------------------------
static inline unsigned cdivu(long a, long b) { return (unsigned)((a + b - 1) / b); }

extern "C" void kernel_launch(void* const* d_in, const int* in_sizes, int n_in,
                              void* d_out, int out_size, void* d_ws, size_t ws_size,
                              hipStream_t stream) {
  (void)in_sizes; (void)n_in; (void)out_size; (void)ws_size;
  const float* feat_d = (const float*)d_in[0];
  const float* feat_z = (const float*)d_in[1];
  const float* att_d  = (const float*)d_in[2];
  const float* att_z  = (const float*)d_in[3];
  const int*   src_d  = (const int*)d_in[4];
  const int*   dst_d  = (const int*)d_in[5];
  const int*   src_z  = (const int*)d_in[6];
  const int*   dst_z  = (const int*)d_in[7];
  const float *dW1 = (const float*)d_in[8],  *db1 = (const float*)d_in[9];
  const float *dW2 = (const float*)d_in[10], *db2 = (const float*)d_in[11];
  const float *dW3 = (const float*)d_in[12], *db3 = (const float*)d_in[13];
  const float *Wq_d = (const float*)d_in[14], *bq_d = (const float*)d_in[15];
  const float *Wk_d = (const float*)d_in[16], *bk_d = (const float*)d_in[17];
  const float *beta_d = (const float*)d_in[18];
  const float *zW1 = (const float*)d_in[19], *zb1 = (const float*)d_in[20];
  const float *zW2 = (const float*)d_in[21], *zb2 = (const float*)d_in[22];
  const float *zW3 = (const float*)d_in[23], *zb3 = (const float*)d_in[24];
  const float *Wq_z = (const float*)d_in[25], *bq_z = (const float*)d_in[26];
  const float *Wk_z = (const float*)d_in[27], *bk_z = (const float*)d_in[28];
  const float *beta_z = (const float*)d_in[29];
  const float *wattn = (const float*)d_in[30];
  const float *Wdec  = (const float*)d_in[31];
  const float *bdec  = (const float*)d_in[32];

  char* ws = (char*)d_ws;
  float* deg_dp = (float*)(ws + OFF_DEG_D);
  float* deg_zp = (float*)(ws + OFF_DEG_Z);
  float* agg_dp = (float*)(ws + OFF_AGG32_D);
  float* agg_zp = (float*)(ws + OFF_AGG32_Z);
  half_t* aggh_d = (half_t*)(ws + OFF_AGGH_D);
  half_t* aggh_z = (half_t*)(ws + OFF_AGGH_Z);
  half_t* h1_d = (half_t*)(ws + OFF_H1_D);
  half_t* h1_z = (half_t*)(ws + OFF_H1_Z);
  half_t* h2_d = (half_t*)(ws + OFF_H2_D);
  half_t* h2_z = (half_t*)(ws + OFF_H2_Z);
  half_t* proj_d = (half_t*)(ws + OFF_PROJ_D);
  half_t* proj_z = (half_t*)(ws + OFF_PROJ_Z);
  half_t* Qd = (half_t*)(ws + OFF_Q_D);
  half_t* Kd = (half_t*)(ws + OFF_K_D);
  half_t* Qz = (half_t*)(ws + OFF_Q_Z);
  half_t* Kz = (half_t*)(ws + OFF_K_Z);
  half_t* outb_d = (half_t*)(ws + OFF_OUT_D);
  half_t* outb_z = (half_t*)(ws + OFF_OUT_Z);
  half_t* Abig = (half_t*)(ws + OFF_ABIG);
  half_t* Bbig = (half_t*)(ws + OFF_BBIG);
  half_t* W1T_d = (half_t*)(ws + OFF_W1T_D);
  half_t* W2T_d = (half_t*)(ws + OFF_W2T_D);
  half_t* W3T_d = (half_t*)(ws + OFF_W3T_D);
  half_t* W1T_z = (half_t*)(ws + OFF_W1T_Z);
  half_t* W2T_z = (half_t*)(ws + OFF_W2T_Z);
  half_t* W3T_z = (half_t*)(ws + OFF_W3T_Z);
  half_t* WQT_d = (half_t*)(ws + OFF_WQT_D);
  half_t* WKT_d = (half_t*)(ws + OFF_WKT_D);
  half_t* WQT_z = (half_t*)(ws + OFF_WQT_Z);
  half_t* WKT_z = (half_t*)(ws + OFF_WKT_Z);
  half_t* WDECT = (half_t*)(ws + OFF_WDECT);

  const int B = 256;

  // --- zero accumulators (deg arrays contiguous; agg32 arrays contiguous) ---
  hipMemsetAsync(ws + OFF_DEG_D, 0, (size_t)P * (ND + NZ) * 4, stream);
  hipMemsetAsync(ws + OFF_AGG32_D, 0, SZ_AGG32, stream);

  // --- SPMM ---
  k_deg<<<cdivu(P * ED, B), B, 0, stream>>>(src_d, deg_dp, P * ED, (int)ED, ND);
  k_deg<<<cdivu(P * EZ, B), B, 0, stream>>>(src_z, deg_zp, P * EZ, (int)EZ, NZ);
  k_inv<<<cdivu((long)P * (ND + NZ), B), B, 0, stream>>>(deg_dp, (long)P * (ND + NZ));
  k_scatter<<<cdivu(P * ED * 4, B), B, 0, stream>>>(src_d, dst_d, att_d, feat_d,
                                                    deg_dp, agg_dp, (int)ED, ND);
  k_scatter<<<cdivu(P * EZ * 4, B), B, 0, stream>>>(src_z, dst_z, att_z, feat_z,
                                                    deg_zp, agg_zp, (int)EZ, NZ);
  k_aggh<<<cdivu((long)M * ND * IN, B), B, 0, stream>>>(feat_d, agg_dp, aggh_d, ND);
  k_aggh<<<cdivu((long)M * NZ * IN, B), B, 0, stream>>>(feat_z, agg_zp, aggh_z, NZ);

  // --- weight transpose-convert to f16 (Kd,Nd)->(Nd,Kd) ---
  k_wT<<<cdivu((long)M * IN * MID, B), B, 0, stream>>>(dW1, W1T_d, M, IN, MID);
  k_wT<<<cdivu((long)M * MID * MID, B), B, 0, stream>>>(dW2, W2T_d, M, MID, MID);
  k_wT<<<cdivu((long)M * MID * HID, B), B, 0, stream>>>(dW3, W3T_d, M, MID, HID);
  k_wT<<<cdivu((long)M * IN * MID, B), B, 0, stream>>>(zW1, W1T_z, M, IN, MID);
  k_wT<<<cdivu((long)M * MID * MID, B), B, 0, stream>>>(zW2, W2T_z, M, MID, MID);
  k_wT<<<cdivu((long)M * MID * HID, B), B, 0, stream>>>(zW3, W3T_z, M, MID, HID);
  k_wT<<<cdivu((long)HID * HID, B), B, 0, stream>>>(Wq_d, WQT_d, 1, HID, HID);
  k_wT<<<cdivu((long)HID * HID, B), B, 0, stream>>>(Wk_d, WKT_d, 1, HID, HID);
  k_wT<<<cdivu((long)HID * HID, B), B, 0, stream>>>(Wq_z, WQT_z, 1, HID, HID);
  k_wT<<<cdivu((long)HID * HID, B), B, 0, stream>>>(Wk_z, WKT_z, 1, HID, HID);
  k_wT<<<cdivu((long)M * HID * HID, B), B, 0, stream>>>(Wdec, WDECT, M, HID, HID);

  dim3 blk(256);
  // --- projector (batched over M channels; BN=64 tiles since MID=192) ---
  // drug
  k_gemm<64, 4, 2, true, true><<<dim3(MID / 64, ND / 128, M), blk, 0, stream>>>(
      aggh_d, W1T_d, db1, h1_d, IN, MID,
      (long)ND * IN, (long)MID * IN, (long)MID, (long)ND * MID);
  k_gemm<64, 4, 2, true, true><<<dim3(MID / 64, ND / 128, M), blk, 0, stream>>>(
      h1_d, W2T_d, db2, h2_d, MID, MID,
      (long)ND * MID, (long)MID * MID, (long)MID, (long)ND * MID);
  k_gemm<64, 4, 2, false, true><<<dim3(HID / 64, ND / 128, M), blk, 0, stream>>>(
      h2_d, W3T_d, db3, proj_d, MID, HID,
      (long)ND * MID, (long)HID * MID, (long)HID, (long)ND * HID);
  // disease
  k_gemm<64, 4, 2, true, true><<<dim3(MID / 64, NZ / 128, M), blk, 0, stream>>>(
      aggh_z, W1T_z, zb1, h1_z, IN, MID,
      (long)NZ * IN, (long)MID * IN, (long)MID, (long)NZ * MID);
  k_gemm<64, 4, 2, true, true><<<dim3(MID / 64, NZ / 128, M), blk, 0, stream>>>(
      h1_z, W2T_z, zb2, h2_z, MID, MID,
      (long)NZ * MID, (long)MID * MID, (long)MID, (long)NZ * MID);
  k_gemm<64, 4, 2, false, true><<<dim3(HID / 64, NZ / 128, M), blk, 0, stream>>>(
      h2_z, W3T_z, zb3, proj_z, MID, HID,
      (long)NZ * MID, (long)HID * MID, (long)HID, (long)NZ * HID);

  // --- Q/K projections (rows = M*n, shared weights; BN=128) ---
  k_gemm<128, 2, 4, false, true><<<dim3(1, M * ND / 128, 1), blk, 0, stream>>>(
      proj_d, WQT_d, bq_d, Qd, HID, HID, 0, 0, 0, 0);
  k_gemm<128, 2, 4, false, true><<<dim3(1, M * ND / 128, 1), blk, 0, stream>>>(
      proj_d, WKT_d, bk_d, Kd, HID, HID, 0, 0, 0, 0);
  k_gemm<128, 2, 4, false, true><<<dim3(1, M * NZ / 128, 1), blk, 0, stream>>>(
      proj_z, WQT_z, bq_z, Qz, HID, HID, 0, 0, 0, 0);
  k_gemm<128, 2, 4, false, true><<<dim3(1, M * NZ / 128, 1), blk, 0, stream>>>(
      proj_z, WKT_z, bk_z, Kz, HID, HID, 0, 0, 0, 0);

  // --- per-node attention (writes (n,M,H) flat order) ---
  k_attn<<<ND, HID, 0, stream>>>(proj_d, Qd, Kd, beta_d, outb_d, ND);
  k_attn<<<NZ, HID, 0, stream>>>(proj_z, Qz, Kz, beta_z, outb_z, NZ);

  // --- fold softmax(weight_attn) into drug side: Abig (ND, 640) ---
  k_abig<<<cdivu((long)ND * KBIG, B), B, 0, stream>>>(outb_d, wattn, Abig);

  // --- decoder dec_z: Bbig[z, m*128 + k] = out_z[m] @ Wdec[m] + bdec[m] ---
  k_gemm<128, 2, 4, false, true><<<dim3(1, NZ / 128, M), blk, 0, stream>>>(
      outb_z, WDECT, bdec, Bbig, HID, KBIG,
      (long)NZ * HID, (long)HID * HID, (long)HID, (long)HID);

  // --- final fused decoder GEMM: out(8192,4096) = Abig @ Bbig^T, K = 640 ---
  k_gemm<128, 2, 4, false, false><<<dim3(NZ / 128, ND / 128, 1), blk, 0, stream>>>(
      Abig, Bbig, nullptr, (float*)d_out, KBIG, NZ, 0, 0, 0, 0);
}
